// GCN1_3444563771409
// MI455X (gfx1250) — matrix operand order repaired and checked by reading, not compiled, not run on hardware
//
#include <hip/hip_runtime.h>

// ---------------------------------------------------------------------------
// Types for CDNA5 WMMA (wave32): v_wmma_f32_16x16x32_bf16
// ---------------------------------------------------------------------------
typedef __attribute__((ext_vector_type(16))) __bf16 bf16x16;
typedef __attribute__((ext_vector_type(8)))  float  f32x8;
typedef __attribute__((ext_vector_type(4)))  unsigned int u32x4;  // clang vector (nontemporal-ok)

struct Bytes32 { uint4 lo; uint4 hi; };   // 32 bytes == one bf16x16 fragment

// Hardware bf16 conversion (RTNE)
__device__ __forceinline__ unsigned short f2bf(float f) {
  __bf16 h = (__bf16)f;
  return __builtin_bit_cast(unsigned short, h);
}
__device__ __forceinline__ unsigned int pack2bf(float a, float b) {
  return (unsigned int)f2bf(a) | ((unsigned int)f2bf(b) << 16);
}
__device__ __forceinline__ float bf2f(unsigned short h) {
  unsigned int u = ((unsigned int)h) << 16;
  return __builtin_bit_cast(float, u);
}

// ---------------------------------------------------------------------------
// Weight packing: fold BN scale into W, swizzle into per-lane WMMA B-fragment
// layout. Element i of a lane's bf16x16 B-fragment is K = k0 + (i&7) + 8*(lane/16) + 16*(i>>3),
// N = n0 + lane%16 (CDNA5 ISA 16-bit B layout). Packed index:
//   Wp[(((t*KS + s)*32 + lane)*16 + i)]   t = n-tile (16 cols), s = k-step (32 K)
// ---------------------------------------------------------------------------
__global__ void pack_w_kernel(const float* __restrict__ W,
                              const float* __restrict__ g,
                              const float* __restrict__ v,
                              unsigned short* __restrict__ Wp,
                              int K, int DN) {
  const int total = K * DN;
  const int idx = blockIdx.x * blockDim.x + threadIdx.x;
  if (idx >= total) return;
  const int KS = K / 32;
  const int i = idx & 15;
  const int l = (idx >> 4) & 31;
  const int ts = idx >> 9;
  const int s = ts % KS;
  const int t = ts / KS;
  const int k = s * 32 + (i & 7) + 8 * (l >> 4) + 16 * (i >> 3);
  const int n = t * 16 + (l & 15);
  float scale = 1.0f;
  if (g) scale = g[n] * rsqrtf(v[n] + 1e-5f);
  Wp[idx] = f2bf(W[(size_t)k * DN + n] * scale);
}

__global__ void fold_bias_kernel(const float* __restrict__ b,
                                 const float* __restrict__ g,
                                 const float* __restrict__ be,
                                 const float* __restrict__ m,
                                 const float* __restrict__ v,
                                 float* __restrict__ out, int DN) {
  const int j = blockIdx.x * blockDim.x + threadIdx.x;
  if (j >= DN) return;
  if (g) {
    const float s = g[j] * rsqrtf(v[j] + 1e-5f);
    out[j] = (b[j] - m[j]) * s + be[j];
  } else {
    out[j] = b[j];
  }
}

// ---------------------------------------------------------------------------
// GEMM: C_bf16[nrows x DN] = A_bf16[nrows x 256] @ Wp + bias  (BN pre-folded)
// Block = 256 threads = 8 waves. Each wave computes a 32 x 64 strip of C:
// 2 A-fragments (rows m0, m0+16) x 4 N-tiles = 8 WMMA accumulators, so every
// B fragment loaded from L2 feeds two WMMAs (load:wmma = 1.5:1).
// K = 256 fully unrolled in 8 steps of 32.
//   MG = 32-row M-groups per block, NG = 64-col groups (DN = NG*64), MG*NG=8.
// ---------------------------------------------------------------------------
template <int MG, int NG>
__global__ __launch_bounds__(256) void gemm_kernel(
    const unsigned short* __restrict__ A, const unsigned short* __restrict__ Wp,
    const float* __restrict__ bias, unsigned short* __restrict__ C, int nrows) {
  constexpr int K = 256;
  constexpr int KS = K / 32;          // 8 k-steps
  constexpr int DN = NG * 64;
  const int lane = threadIdx.x & 31;
  const int wave = threadIdx.x >> 5;
  const int mg   = wave % MG;
  const int ng   = wave / MG;
  const int m0   = (blockIdx.x * MG + mg) * 32;
  if (m0 >= nrows) return;            // wave-uniform; EXEC stays all-ones
  const int half = lane >> 4;
  const int lrow = lane & 15;
  const int arow0 = min(m0 + lrow, nrows - 1);
  const int arow1 = min(m0 + 16 + lrow, nrows - 1);

  f32x8 acc[2][4] = {};

#pragma unroll
  for (int s = 0; s < KS; ++s) {
    // ---- two A fragments (16x32 bf16): per-lane two contiguous 16B chunks ----
    const unsigned short* ap0 = A + (size_t)arow0 * K + s * 32;
    const unsigned short* ap1 = A + (size_t)arow1 * K + s * 32;
    Bytes32 ab0, ab1;
    ab0.lo = *(const uint4*)(ap0 + 8 * half);
    ab0.hi = *(const uint4*)(ap0 + 16 + 8 * half);
    ab1.lo = *(const uint4*)(ap1 + 8 * half);
    ab1.hi = *(const uint4*)(ap1 + 16 + 8 * half);
    const bf16x16 a0 = __builtin_bit_cast(bf16x16, ab0);
    const bf16x16 a1 = __builtin_bit_cast(bf16x16, ab1);

#pragma unroll
    for (int j = 0; j < 4; ++j) {
      const int t = ng * 4 + j;
      const Bytes32* bp =
          (const Bytes32*)(Wp + ((size_t)(t * KS + s) * 32 + lane) * 16);
      const bf16x16 bfrag = __builtin_bit_cast(bf16x16, *bp);
      acc[0][j] = __builtin_amdgcn_wmma_f32_16x16x32_bf16(
          false, a0, false, bfrag, (short)0, acc[0][j], false, false);
      acc[1][j] = __builtin_amdgcn_wmma_f32_16x16x32_bf16(
          false, a1, false, bfrag, (short)0, acc[1][j], false, false);
    }
  }

  // ---- epilogue: + folded bias, store bf16 (C layout: M = r + 8*half) ----
#pragma unroll
  for (int mf = 0; mf < 2; ++mf) {
#pragma unroll
    for (int j = 0; j < 4; ++j) {
      const int coln = (ng * 4 + j) * 16 + lrow;
      const float bv = bias[coln];
#pragma unroll
      for (int r = 0; r < 8; ++r) {
        const int rown = m0 + 16 * mf + 8 * half + r;
        if (rown < nrows)
          C[(size_t)rown * DN + coln] = f2bf(acc[mf][j][r] + bv);
      }
    }
  }
}

// ---------------------------------------------------------------------------
// SpMM: Y[row[e], :] += w[e] * H[col[e], :]   (H bf16 L2-resident, Y fp32)
// One wave per edge; lane handles CPL contiguous columns (D = CPL*32).
// Gather uses non-temporal loads: 51MB working set lives in L2, not WGP$.
// ---------------------------------------------------------------------------
template <int CPL>
__global__ __launch_bounds__(256) void spmm_kernel(
    const int* __restrict__ erow, const int* __restrict__ ecol,
    const float* __restrict__ ew, const unsigned short* __restrict__ H,
    float* __restrict__ Y, int E) {
  constexpr int D = CPL * 32;
  const int e = blockIdx.x * 8 + (threadIdx.x >> 5);
  if (e >= E) return;
  const int lane = threadIdx.x & 31;
  const int r = erow[e];
  const int c = ecol[e];
  const float wt = ew[e];
  const unsigned short* hp = H + (size_t)c * D + lane * CPL;
  float* yp = Y + (size_t)r * D + lane * CPL;
  if constexpr (CPL == 8) {
    const u32x4 hv = __builtin_nontemporal_load((const u32x4*)hp);
#pragma unroll
    for (int q = 0; q < 4; ++q) {
      const unsigned int u = hv[q];
      unsafeAtomicAdd(yp + 2 * q + 0, wt * bf2f((unsigned short)(u & 0xFFFFu)));
      unsafeAtomicAdd(yp + 2 * q + 1, wt * bf2f((unsigned short)(u >> 16)));
    }
  } else {
    const unsigned int u = __builtin_nontemporal_load((const unsigned int*)hp);
    unsafeAtomicAdd(yp + 0, wt * bf2f((unsigned short)(u & 0xFFFFu)));
    unsafeAtomicAdd(yp + 1, wt * bf2f((unsigned short)(u >> 16)));
  }
}

__global__ void zero_kernel(float4* __restrict__ p, long n4) {
  const long i = (long)blockIdx.x * blockDim.x + threadIdx.x;
  if (i < n4) p[i] = make_float4(0.f, 0.f, 0.f, 0.f);
}

// (optional ReLU) + f32 -> bf16 pack; feeds the GEMM A operand
template <bool RELU>
__global__ void cvt_kernel(const float4* __restrict__ in,
                           uint2* __restrict__ out, long n4) {
  const long i = (long)blockIdx.x * blockDim.x + threadIdx.x;
  if (i >= n4) return;
  float4 v = in[i];
  if constexpr (RELU) {
    v.x = fmaxf(v.x, 0.f); v.y = fmaxf(v.y, 0.f);
    v.z = fmaxf(v.z, 0.f); v.w = fmaxf(v.w, 0.f);
  }
  uint2 o;
  o.x = pack2bf(v.x, v.y);
  o.y = pack2bf(v.z, v.w);
  out[i] = o;
}

// log_softmax over D_OUT=64; one wave per row, 2 elements per lane
__global__ __launch_bounds__(256) void lsm_kernel(const float* __restrict__ Y,
                                                  float* __restrict__ out,
                                                  int nrows) {
  const int row = blockIdx.x * 8 + (threadIdx.x >> 5);
  if (row >= nrows) return;
  const int lane = threadIdx.x & 31;
  const float2 v = *(const float2*)(Y + (size_t)row * 64 + lane * 2);
  float mx = fmaxf(v.x, v.y);
#pragma unroll
  for (int off = 16; off > 0; off >>= 1) mx = fmaxf(mx, __shfl_xor(mx, off, 32));
  float se = __expf(v.x - mx) + __expf(v.y - mx);
#pragma unroll
  for (int off = 16; off > 0; off >>= 1) se += __shfl_xor(se, off, 32);
  const float ls = __logf(se);
  float* op = out + (size_t)row * 64 + lane * 2;
  op[0] = v.x - mx - ls;
  op[1] = v.y - mx - ls;
}

// ---------------------------------------------------------------------------
// Host-side orchestration
// ---------------------------------------------------------------------------
static inline size_t align256(size_t x) { return (x + 255) & ~(size_t)255; }
static inline int cdiv(long a, long b) { return (int)((a + b - 1) / b); }

extern "C" void kernel_launch(void* const* d_in, const int* in_sizes, int n_in,
                              void* d_out, int out_size, void* d_ws, size_t ws_size,
                              hipStream_t stream) {
  const float* x   = (const float*)d_in[0];
  const int* erow  = (const int*)d_in[1];
  const int* ecol  = (const int*)d_in[2];
  const float* ew  = (const float*)d_in[3];
  const float* W1  = (const float*)d_in[4];
  const float* b1  = (const float*)d_in[5];
  const float* g1  = (const float*)d_in[6];
  const float* be1 = (const float*)d_in[7];
  const float* m1  = (const float*)d_in[8];
  const float* v1  = (const float*)d_in[9];
  const float* W2  = (const float*)d_in[10];
  const float* b2  = (const float*)d_in[11];
  const float* g2  = (const float*)d_in[12];
  const float* be2 = (const float*)d_in[13];
  const float* m2  = (const float*)d_in[14];
  const float* v2  = (const float*)d_in[15];
  const float* W3  = (const float*)d_in[16];
  const float* b3  = (const float*)d_in[17];

  const int N = in_sizes[0] / 256;   // 100000
  const int E = in_sizes[1];         // 3200000

  char* ws = (char*)d_ws;
  const size_t oW1 = 0;
  const size_t oW2 = align256(oW1 + 256 * 256 * 2);
  const size_t oW3 = align256(oW2 + 256 * 256 * 2);
  const size_t oB1 = align256(oW3 + 256 * 64 * 2);
  const size_t oB2 = align256(oB1 + 256 * 4);
  const size_t oB3 = align256(oB2 + 256 * 4);
  const size_t oH  = align256(oB3 + 64 * 4);
  const size_t oY  = align256(oH + (size_t)N * 256 * 2);
  const size_t oA  = align256(oY + (size_t)N * 256 * 4);

  unsigned short* Wp1 = (unsigned short*)(ws + oW1);
  unsigned short* Wp2 = (unsigned short*)(ws + oW2);
  unsigned short* Wp3 = (unsigned short*)(ws + oW3);
  float* bias1 = (float*)(ws + oB1);
  float* bias2 = (float*)(ws + oB2);
  float* bias3 = (float*)(ws + oB3);
  unsigned short* hbf  = (unsigned short*)(ws + oH);  // GEMM out (bf16)
  float*          yacc = (float*)(ws + oY);           // SpMM accumulator (f32)
  unsigned short* abf  = (unsigned short*)(ws + oA);  // GEMM A operand (bf16)

  // ---- weight prep (BN folded, fragment-swizzled) ----
  pack_w_kernel<<<cdiv(256 * 256, 256), 256, 0, stream>>>(W1, g1, v1, Wp1, 256, 256);
  pack_w_kernel<<<cdiv(256 * 256, 256), 256, 0, stream>>>(W2, g2, v2, Wp2, 256, 256);
  pack_w_kernel<<<cdiv(256 * 64, 256), 256, 0, stream>>>(W3, nullptr, nullptr, Wp3, 256, 64);
  fold_bias_kernel<<<1, 256, 0, stream>>>(b1, g1, be1, m1, v1, bias1, 256);
  fold_bias_kernel<<<1, 256, 0, stream>>>(b2, g2, be2, m2, v2, bias2, 256);
  fold_bias_kernel<<<1, 64, 0, stream>>>(b3, nullptr, nullptr, nullptr, nullptr, bias3, 64);

  const long n4_256 = (long)N * 64;   // N*256 floats / 4
  const long n4_64  = (long)N * 16;   // N*64 floats / 4

  // ---- layer 1 ----
  cvt_kernel<false><<<cdiv(n4_256, 256), 256, 0, stream>>>((const float4*)x, (uint2*)abf, n4_256);
  gemm_kernel<2, 4><<<cdiv(N, 64), 256, 0, stream>>>(abf, Wp1, bias1, hbf, N);
  zero_kernel<<<cdiv(n4_256, 256), 256, 0, stream>>>((float4*)yacc, n4_256);
  spmm_kernel<8><<<cdiv(E, 8), 256, 0, stream>>>(erow, ecol, ew, hbf, yacc, E);
  cvt_kernel<true><<<cdiv(n4_256, 256), 256, 0, stream>>>((const float4*)yacc, (uint2*)abf, n4_256);

  // ---- layer 2 ----
  gemm_kernel<2, 4><<<cdiv(N, 64), 256, 0, stream>>>(abf, Wp2, bias2, hbf, N);
  zero_kernel<<<cdiv(n4_256, 256), 256, 0, stream>>>((float4*)yacc, n4_256);
  spmm_kernel<8><<<cdiv(E, 8), 256, 0, stream>>>(erow, ecol, ew, hbf, yacc, E);
  cvt_kernel<true><<<cdiv(n4_256, 256), 256, 0, stream>>>((const float4*)yacc, (uint2*)abf, n4_256);

  // ---- output layer (DN = 64) ----
  gemm_kernel<8, 1><<<cdiv(N, 256), 256, 0, stream>>>(abf, Wp3, bias3, hbf, N);
  zero_kernel<<<cdiv(n4_64, 256), 256, 0, stream>>>((float4*)yacc, n4_64);
  spmm_kernel<2><<<cdiv(E, 8), 256, 0, stream>>>(erow, ecol, ew, hbf, yacc, E);
  lsm_kernel<<<cdiv(N, 8), 256, 0, stream>>>(yacc, (float*)d_out, N);
}
